// QGRUBlock_25589415150153
// MI455X (gfx1250) — matrix operand order repaired and checked by reading, not compiled, not run on hardware
//
#include <hip/hip_runtime.h>

// ---------------------------------------------------------------------------
// QGRU block for MI455X (gfx1250, wave32, WMMA + async global->LDS DMA).
//   B=4, L=4096, D=1024, H=2048
// Six bf16 WMMA GEMMs + sequential recurrence scan + fused epilogues.
// GEMM: 256x128x32 block tile, 8 waves (4Mx2N), wave tile 64x64 = 4x4 WMMA
// fragments -> 16 v_wmma per K-step per wave, 1.0 WMMA per ds_load_b128.
// Tile staging via GLOBAL_LOAD_ASYNC_TO_LDS_B128 (ASYNCcnt): zero VGPR
// staging. Epilogue is templated on MODE (straight-line) and the kernel is
// budgeted for 2 waves/EU so accumulators never spill.
// ---------------------------------------------------------------------------

typedef unsigned short u16;
typedef __attribute__((ext_vector_type(16))) __bf16 v16bf;
typedef __attribute__((ext_vector_type(8)))  float  v8f;

#define BMT 256
#define BNT 128
#define BKT 32
#define LDSS 40   // u16 per LDS row = 80 bytes: conflict-friendly, 16B-aligned

enum GemmMode {
  MODE_SIG_DECAY = 0,   // outH = bf16( sigmoid(z) * (col * decay_scale) )
  MODE_TANH      = 1,   // outH = bf16( tanh(z) )
  MODE_PLAIN_H   = 2,   // outH = bf16( z )
  MODE_SWIGLU_H  = 3,   // outH = bf16( bf2f(aux) * silu(z) )
  MODE_RES_F     = 4    // outF = z + res
};

union Frag16 { v16bf v; uint4 q[2]; };

__device__ __forceinline__ u16 f2bf(float f) {
  unsigned u = __float_as_uint(f);
  unsigned r = 0x7FFFu + ((u >> 16) & 1u);   // round-to-nearest-even
  return (u16)((u + r) >> 16);
}
__device__ __forceinline__ float bf2f(u16 h) {
  return __uint_as_float(((unsigned)h) << 16);
}
__device__ __forceinline__ float sigmoidf_(float z) {
  return 1.f / (1.f + __expf(-z));
}

// --- CDNA5 async global->LDS DMA (16B per lane), tracked by ASYNCcnt -------
// LDS aperture uses addr[31:0] directly (ISA 10.2), so the low 32 bits of a
// flat __shared__ pointer are the LDS byte offset for the VDST operand.
__device__ __forceinline__ void async_copy_b128(unsigned ldsOff, const u16* g) {
  asm volatile("global_load_async_to_lds_b128 %0, %1, off"
               :: "v"(ldsOff), "v"(g) : "memory");
}
__device__ __forceinline__ void wait_async0() {
  asm volatile("s_wait_asynccnt 0" ::: "memory");
}

// ---------------------------------------------------------------------------
// f32 -> bf16 conversion (weights / activations), grid-stride
// ---------------------------------------------------------------------------
__global__ __launch_bounds__(256)
void f32_to_bf16_kernel(const float* __restrict__ in, u16* __restrict__ out, long n) {
  long i = (long)blockIdx.x * blockDim.x + threadIdx.x;
  long stride = (long)gridDim.x * blockDim.x;
  for (; i < n; i += stride) out[i] = f2bf(in[i]);
}

// ---------------------------------------------------------------------------
// RMSNorm: one block (256 threads) per row of D=1024; writes bf16.
// ---------------------------------------------------------------------------
__global__ __launch_bounds__(256)
void rmsnorm_bf16_kernel(const float* __restrict__ x, const float* __restrict__ g,
                         u16* __restrict__ out, int D, float eps) {
  __shared__ float red[8];
  const int row = blockIdx.x;
  const int tid = threadIdx.x;
  const float4* xr = (const float4*)(x + (size_t)row * D);
  float4 v = xr[tid];
  float ss = v.x * v.x + v.y * v.y + v.z * v.z + v.w * v.w;
#pragma unroll
  for (int o = 16; o > 0; o >>= 1) ss += __shfl_xor(ss, o, 32);
  if ((tid & 31) == 0) red[tid >> 5] = ss;
  __syncthreads();
  float tot = red[0] + red[1] + red[2] + red[3] + red[4] + red[5] + red[6] + red[7];
  float scale = rsqrtf(tot / (float)D + eps);
  const float4* gr = (const float4*)g;
  float4 gv = gr[tid];
  uint2 p;
  p.x = (unsigned)f2bf(v.x * scale * gv.x) | ((unsigned)f2bf(v.y * scale * gv.y) << 16);
  p.y = (unsigned)f2bf(v.z * scale * gv.z) | ((unsigned)f2bf(v.w * scale * gv.w) << 16);
  *(uint2*)(out + (size_t)row * D + tid * 4) = p;
}

// ---------------------------------------------------------------------------
// WMMA GEMM:  C[M,N] = epilogue( A[M,K] (bf16, row-major) x W[N,K]^T (bf16) )
// Block tile 256x128x32, 8 waves (4Mx2N), wave tile 64x64 = 4x4 WMMA 16x16x32.
// Double-buffered LDS (80B padded rows) filled by async global->LDS DMA.
// ---------------------------------------------------------------------------
template <int MODE>
__global__ __launch_bounds__(256)
__attribute__((amdgpu_waves_per_eu(2)))
void gemm_bf16_kernel(const u16* __restrict__ A, const u16* __restrict__ W,
                      const float* __restrict__ bias,
                      int M, int N, int K,
                      float* __restrict__ outF, u16* __restrict__ outH,
                      const float* __restrict__ res, const u16* __restrict__ aux,
                      float decay_scale) {
  __shared__ __align__(16) u16 lA[2][BMT * LDSS];   // 2 x 20 KB
  __shared__ __align__(16) u16 lW[2][BNT * LDSS];   // 2 x 10 KB

  const int tid   = threadIdx.x;
  const int lane  = tid & 31;
  const int wid   = tid >> 5;
  const int waveM = wid >> 1;       // 0..3
  const int waveN = wid & 1;        // 0..1
  const int lrow  = lane & 15;
  const int lhalf = lane >> 4;      // 0/1 -> K-half selector

  const int bm = blockIdx.y * BMT;
  const int bn = blockIdx.x * BNT;
  const int nk = K / BKT;

  // Staging: A tile = 256 rows x 4 chunks = 1024 16B-chunks (4 per thread);
  //          W tile = 128 rows x 4 chunks =  512 16B-chunks (2 per thread).
  int ar[4], ac[4], wr[2], wc[2];
#pragma unroll
  for (int j = 0; j < 4; ++j) { int c = tid + j * 256; ar[j] = c >> 2; ac[j] = (c & 3) * 8; }
#pragma unroll
  for (int j = 0; j < 2; ++j) { int c = tid + j * 256; wr[j] = c >> 2; wc[j] = (c & 3) * 8; }

  const unsigned aBase = (unsigned)(uintptr_t)(&lA[0][0]);
  const unsigned wBase = (unsigned)(uintptr_t)(&lW[0][0]);

  // issue one stage of async DMA into LDS buffer `buf` for K-offset k0
  auto issue_stage = [&](int buf, int k0) {
    const unsigned aOff = aBase + (unsigned)(buf * BMT * LDSS) * 2u;
    const unsigned wOff = wBase + (unsigned)(buf * BNT * LDSS) * 2u;
#pragma unroll
    for (int j = 0; j < 4; ++j)
      async_copy_b128(aOff + (unsigned)(ar[j] * LDSS + ac[j]) * 2u,
                      A + (size_t)(bm + ar[j]) * K + k0 + ac[j]);
#pragma unroll
    for (int j = 0; j < 2; ++j)
      async_copy_b128(wOff + (unsigned)(wr[j] * LDSS + wc[j]) * 2u,
                      W + (size_t)(bn + wr[j]) * K + k0 + wc[j]);
  };

  // --- preload stage 0 ---
  issue_stage(0, 0);
  wait_async0();
  __syncthreads();

  v8f acc[4][4] = {};

  for (int kk = 0; kk < nk; ++kk) {
    const int cur = kk & 1;
    const bool pf = (kk + 1 < nk);
    if (pf) issue_stage(1 - cur, (kk + 1) * BKT);   // DMA overlaps compute below

    // A fragments: ISA 16-bit A 16x32 layout.
    // lane<16: K 0..7 in q0, 16..23 in q1; lane>=16: K 8..15 / 24..31.
    Frag16 fa[4];
#pragma unroll
    for (int mt = 0; mt < 4; ++mt) {
      const u16* p = &lA[cur][(waveM * 64 + mt * 16 + lrow) * LDSS + lhalf * 8];
      fa[mt].q[0] = *(const uint4*)p;
      fa[mt].q[1] = *(const uint4*)(p + 16);
    }
    // B fragments: lane = N column; lane<16: K 0..7 / 16..23, lane>=16: K 8..15 / 24..31.
    Frag16 fw[4];
#pragma unroll
    for (int nt = 0; nt < 4; ++nt) {
      const u16* p = &lW[cur][(waveN * 64 + nt * 16 + lrow) * LDSS + lhalf * 8];
      fw[nt].q[0] = *(const uint4*)p;
      fw[nt].q[1] = *(const uint4*)(p + 16);
    }

#pragma unroll
    for (int mt = 0; mt < 4; ++mt)
#pragma unroll
      for (int nt = 0; nt < 4; ++nt)
        acc[mt][nt] = __builtin_amdgcn_wmma_f32_16x16x32_bf16(
            false, fa[mt].v, false, fw[nt].v, (short)0, acc[mt][nt], false, false);

    if (pf) wait_async0();   // DMA for next buffer complete before barrier
    __syncthreads();
  }

  // --- epilogue (straight-line, single MODE) ---
  // C fragment layout: element (vgpr r, lane l): row += r + 8*(l>>4), col = l&15.
#pragma unroll
  for (int mt = 0; mt < 4; ++mt) {
#pragma unroll
    for (int nt = 0; nt < 4; ++nt) {
      const int col = bn + waveN * 64 + nt * 16 + lrow;
      const float b = bias[col];
      const int rowBase = bm + waveM * 64 + mt * 16 + lhalf * 8;
#pragma unroll
      for (int r = 0; r < 8; ++r) {
        const size_t idx = (size_t)(rowBase + r) * N + col;
        const float z = acc[mt][nt][r] + b;
        if (MODE == MODE_SIG_DECAY) {
          outH[idx] = f2bf(sigmoidf_(z) * ((float)col * decay_scale));
        } else if (MODE == MODE_TANH) {
          outH[idx] = f2bf(tanhf(z));
        } else if (MODE == MODE_PLAIN_H) {
          outH[idx] = f2bf(z);
        } else if (MODE == MODE_SWIGLU_H) {
          outH[idx] = f2bf(bf2f(aux[idx]) * (z * sigmoidf_(z)));
        } else { // MODE_RES_F
          outF[idx] = z + res[idx];
        }
      }
    }
  }
}

// ---------------------------------------------------------------------------
// Sequential QGRU recurrence: one lane per (b,h) channel, coalesced along H.
//   h[t] = (1-r[t])*h[t-1] + r[t]*i[t]     (h[-1] = hidden)
// ---------------------------------------------------------------------------
__global__ __launch_bounds__(256)
void qgru_scan_kernel(const u16* __restrict__ gR, const u16* __restrict__ gI,
                      const float* __restrict__ hidden, u16* __restrict__ hOut,
                      float* __restrict__ hLast, int L, int H) {
  const int ch = blockIdx.x * blockDim.x + threadIdx.x;  // [0, B*H)
  const int b  = ch / H;
  const int hc = ch - b * H;
  float hv = hidden[ch];
  const size_t base = (size_t)b * L * H + hc;
#pragma unroll 4
  for (int t = 0; t < L; ++t) {
    const size_t off = base + (size_t)t * H;
    const float rr = bf2f(gR[off]);
    const float ii = bf2f(gI[off]);
    hv = (1.f - rr) * hv + rr * ii;
    hOut[off] = f2bf(hv);
  }
  hLast[ch] = hv;
}

// ---------------------------------------------------------------------------
extern "C" void kernel_launch(void* const* d_in, const int* in_sizes, int n_in,
                              void* d_out, int out_size, void* d_ws, size_t ws_size,
                              hipStream_t stream) {
  (void)in_sizes; (void)n_in; (void)out_size; (void)ws_size;
  const int B = 4, L = 4096, D = 1024, H = 2048;
  const size_t MBL = (size_t)B * L;  // 16384 rows

  const float* x        = (const float*)d_in[0];
  const float* hidden   = (const float*)d_in[1];
  const float* w_forget = (const float*)d_in[2];
  const float* b_forget = (const float*)d_in[3];
  const float* w_input  = (const float*)d_in[4];
  const float* b_input  = (const float*)d_in[5];
  const float* w_hout   = (const float*)d_in[6];
  const float* b_hout   = (const float*)d_in[7];
  const float* w_fc     = (const float*)d_in[8];
  const float* b_fc     = (const float*)d_in[9];
  const float* w_fc_act = (const float*)d_in[10];
  const float* b_fc_act = (const float*)d_in[11];
  const float* w_fout   = (const float*)d_in[12];
  const float* b_fout   = (const float*)d_in[13];
  const float* g_norm1  = (const float*)d_in[14];
  const float* g_norm2  = (const float*)d_in[15];

  float* out   = (float*)d_out;
  float* hLast = out + MBL * D;   // second tuple output: h[:, -1, :]

  // ---- workspace carve-up (all 256B aligned; total ~326 MB) ----
  char* ws = (char*)d_ws;
  size_t off = 0;
  auto carve = [&](size_t bytes) -> void* {
    void* p = ws + off;
    off += (bytes + 255) & ~(size_t)255;
    return p;
  };
  u16* wF_h   = (u16*)carve((size_t)H * D * 2);
  u16* wI_h   = (u16*)carve((size_t)H * D * 2);
  u16* wHo_h  = (u16*)carve((size_t)D * H * 2);
  u16* wFc_h  = (u16*)carve((size_t)H * D * 2);
  u16* wAc_h  = (u16*)carve((size_t)H * D * 2);
  u16* wFo_h  = (u16*)carve((size_t)D * H * 2);
  u16* xn_h   = (u16*)carve(MBL * D * 2);       // rmsnorm output (reused for xn2)
  u16* gR_h   = (u16*)carve(MBL * H * 2);       // remember   (reused for z_fc)
  u16* gI_h   = (u16*)carve(MBL * H * 2);       // tanh input (reused for ff)
  u16* h_h    = (u16*)carve(MBL * H * 2);       // recurrence states (bf16)
  float* x1   = (float*)carve(MBL * D * 4);     // residual stream after QGRU

  const dim3 blk(256);

  // 1) weights f32 -> bf16
  {
    const long nW = (long)H * D;
    f32_to_bf16_kernel<<<1024, blk, 0, stream>>>(w_forget, wF_h, nW);
    f32_to_bf16_kernel<<<1024, blk, 0, stream>>>(w_input,  wI_h, nW);
    f32_to_bf16_kernel<<<1024, blk, 0, stream>>>(w_hout,   wHo_h, nW);
    f32_to_bf16_kernel<<<1024, blk, 0, stream>>>(w_fc,     wFc_h, nW);
    f32_to_bf16_kernel<<<1024, blk, 0, stream>>>(w_fc_act, wAc_h, nW);
    f32_to_bf16_kernel<<<1024, blk, 0, stream>>>(w_fout,   wFo_h, nW);
  }

  // 2) xn = rmsnorm(x, g_norm1)  -> bf16
  rmsnorm_bf16_kernel<<<(unsigned)MBL, blk, 0, stream>>>(x, g_norm1, xn_h, D, 1e-6f);

  // 3) remember = sigmoid(xn @ wF^T + b) * decay[n]
  {
    dim3 grid(H / BNT, (unsigned)(MBL / BMT));
    gemm_bf16_kernel<MODE_SIG_DECAY><<<grid, blk, 0, stream>>>(xn_h, wF_h, b_forget,
        (int)MBL, H, D, nullptr, gR_h, nullptr, nullptr, 1.0f / (float)(H - 1));
  }
  // 4) inp = tanh(xn @ wI^T + b)
  {
    dim3 grid(H / BNT, (unsigned)(MBL / BMT));
    gemm_bf16_kernel<MODE_TANH><<<grid, blk, 0, stream>>>(xn_h, wI_h, b_input,
        (int)MBL, H, D, nullptr, gI_h, nullptr, nullptr, 0.f);
  }
  // 5) sequential recurrence -> h (bf16) and h_last (f32, output tail)
  qgru_scan_kernel<<<(B * H) / 256, blk, 0, stream>>>(gR_h, gI_h, hidden, h_h, hLast, L, H);

  // 6) x1 = h @ w_hout^T + b_hout + x
  {
    dim3 grid(D / BNT, (unsigned)(MBL / BMT));
    gemm_bf16_kernel<MODE_RES_F><<<grid, blk, 0, stream>>>(h_h, wHo_h, b_hout,
        (int)MBL, D, H, x1, nullptr, x, nullptr, 0.f);
  }
  // 7) xn2 = rmsnorm(x1, g_norm2)  (reuse xn buffer)
  rmsnorm_bf16_kernel<<<(unsigned)MBL, blk, 0, stream>>>(x1, g_norm2, xn_h, D, 1e-6f);

  // 8) z_fc = xn2 @ w_fc^T + b_fc   (reuse gR buffer)
  {
    dim3 grid(H / BNT, (unsigned)(MBL / BMT));
    gemm_bf16_kernel<MODE_PLAIN_H><<<grid, blk, 0, stream>>>(xn_h, wFc_h, b_fc,
        (int)MBL, H, D, nullptr, gR_h, nullptr, nullptr, 0.f);
  }
  // 9) ff = z_fc * silu(xn2 @ w_fc_act^T + b)   (reuse gI buffer)
  {
    dim3 grid(H / BNT, (unsigned)(MBL / BMT));
    gemm_bf16_kernel<MODE_SWIGLU_H><<<grid, blk, 0, stream>>>(xn_h, wAc_h, b_fc_act,
        (int)MBL, H, D, nullptr, gI_h, nullptr, gR_h, 0.f);
  }
  // 10) out = ff @ w_fout^T + b_fout + x1
  {
    dim3 grid(D / BNT, (unsigned)(MBL / BMT));
    gemm_bf16_kernel<MODE_RES_F><<<grid, blk, 0, stream>>>(gI_h, wFo_h, b_fout,
        (int)MBL, D, H, out, nullptr, x1, nullptr, 0.f);
  }
}
